// NewEncoder_29875792511308
// MI455X (gfx1250) — matrix operand order repaired and checked by reading, not compiled, not run on hardware
//
#include <hip/hip_runtime.h>

typedef __attribute__((ext_vector_type(16))) __bf16   v16bf;
typedef __attribute__((ext_vector_type(8)))  __bf16   v8bf;
typedef __attribute__((ext_vector_type(8)))  float    v8f;
typedef __attribute__((ext_vector_type(4)))  unsigned u32x4;
typedef __attribute__((ext_vector_type(8)))  int      i32x8;
typedef __attribute__((ext_vector_type(4)))  int      i32x4;

#define FDIM 128
#define WPITCH 136   // bf16 elements per row (272 B): 16B-aligned fragments, bank-conflict-free
#define WTILE_BYTES (FDIM * WPITCH * 2)          // 34816
#define WTILE_DW    (WTILE_BYTES / 4)            // 8704 (fits 16-bit tile_dim0)

__device__ __forceinline__ __bf16 f2bf(float f) {
  unsigned u = __builtin_bit_cast(unsigned, f);
  unsigned r = (u + 0x7FFFu + ((u >> 16) & 1u)) >> 16;  // round-to-nearest-even
  return __builtin_bit_cast(__bf16, (unsigned short)r);
}

__global__ void k_zero(float* __restrict__ p, int n) {
  int i = blockIdx.x * blockDim.x + threadIdx.x;
  if (i < n) p[i] = 0.0f;
}

__global__ void k_deg(const int* __restrict__ snd, float* __restrict__ deg, int E) {
  int e = blockIdx.x * blockDim.x + threadIdx.x;
  if (e < E) atomicAdd(&deg[snd[e]], 1.0f);
}

__global__ void k_dinv(float* __restrict__ d, int n) {
  int i = blockIdx.x * blockDim.x + threadIdx.x;
  if (i < n) d[i] = rsqrtf(d[i] + 1.0f);  // +1 = appended self loop
}

// Convert W(128x128 f32, row major) -> bf16, transposed + padded: Wt[n*WPITCH + k]
__global__ void k_wconv(const float* __restrict__ W, __bf16* __restrict__ Wt) {
  int i = blockIdx.x * blockDim.x + threadIdx.x;
  if (i < FDIM * FDIM) {
    int k = i >> 7, n = i & 127;
    Wt[(size_t)n * WPITCH + k] = f2bf(W[i]);
  }
}

// Y = A(N x 128) @ W(128 x 128)  [+ Z]
// Wg: pre-converted bf16 transposed W in global. Each block DMAs it to LDS via TDM.
// Wave w computes the 16x16 tile at (blockIdx.x*16, w*16).
__global__ void __launch_bounds__(256)
k_gemm_wmma(const float* __restrict__ A, const __bf16* __restrict__ Wg,
            const float* __restrict__ Z, float* __restrict__ Y, int nrows) {
  __shared__ __bf16 Wt[FDIM * WPITCH];   // ~34 KB

  if (threadIdx.x < 32) {                // wave 0 issues one TDM load for the block
    unsigned lds_off = (unsigned)(size_t)(void*)&Wt[0];
    unsigned long long ga = (unsigned long long)(size_t)Wg;
    // D# group0: count=1, lds_addr, global_addr[56:0], type=2
    u32x4 g0 = { 1u, lds_off, (unsigned)ga,
                 (unsigned)((ga >> 32) & 0x01FFFFFFull) | (2u << 30) };
    // D# group1: data_size=4B; tensor_dim0 = tile_dim0 = dim0_stride = 8704 dwords; 1-D tile
    i32x8 g1 = { (int)(2u << 16),
                 (int)((unsigned)(WTILE_DW & 0xFFFF) << 16),   // tensor_dim0 lo -> bits[63:48]
                 (int)(WTILE_DW >> 16),                        // tensor_dim0 hi -> bits[79:64]
                 (int)((unsigned)(WTILE_DW & 0xFFFF) << 16),   // tile_dim0     -> bits[127:112]
                 0,                                            // tile_dim1/2 = 0 (unused)
                 (int)WTILE_DW,                                // tensor_dim0_stride lo
                 0, 0 };
    i32x4 gz = { 0, 0, 0, 0 };
#if defined(__clang_major__) && (__clang_major__ >= 23)
    i32x8 gz8 = { 0, 0, 0, 0, 0, 0, 0, 0 };
    __builtin_amdgcn_tensor_load_to_lds(g0, g1, gz, gz, gz8, 0);
#else
    __builtin_amdgcn_tensor_load_to_lds(g0, g1, gz, gz, 0);
#endif
    __builtin_amdgcn_s_wait_tensorcnt(0);
  }
  __syncthreads();

  const int wave = threadIdx.x >> 5;
  const int lane = threadIdx.x & 31;
  const int hi   = lane >> 4;          // half-wave select
  const int lm   = lane & 15;
  const int m0   = blockIdx.x << 4;    // row tile base
  const int ncol = (wave << 4) + lm;   // this lane's output column
  const int mrow = m0 + lm;            // this lane's A row
  // Clamp instead of branching: a clamped A row only feeds D rows that are never stored.
  const float* abase = A + (size_t)((mrow < nrows) ? mrow : 0) * FDIM;

  v8f acc = {};
  #pragma unroll
  for (int kt = 0; kt < FDIM; kt += 32) {
    // A 16x32 bf16 layout: lanes 0-15 K=[0..15]; lanes 16-31 K=[16..31]
    const float* ap = abase + kt + hi * 16;
    v16bf a;
    #pragma unroll
    for (int j = 0; j < 8; ++j) a[j] = f2bf(ap[j]);
    #pragma unroll
    for (int j = 0; j < 8; ++j) a[8 + j] = f2bf(ap[8 + j]);
    // B 32x16 bf16 layout: col = lane%16, K = hi*16 + j  -> contiguous in Wt
    const __bf16* wl = &Wt[ncol * WPITCH + kt + hi * 16];
    const v8bf b0 = *(const v8bf*)(wl);      // 16B-aligned ds_load_b128
    const v8bf b1 = *(const v8bf*)(wl + 8);
    v16bf b;
    #pragma unroll
    for (int j = 0; j < 8; ++j) { b[j] = b0[j]; b[8 + j] = b1[j]; }
    acc = __builtin_amdgcn_wmma_f32_16x16x32_bf16(false, a, false, b,
                                                  (short)0, acc, false, false);
  }
  // D layout: VGPR v -> row m0 + v + 8*hi, col = ncol
  #pragma unroll
  for (int v = 0; v < 8; ++v) {
    int r = m0 + v + (hi << 3);
    if (r < nrows) {
      size_t o = (size_t)r * FDIM + ncol;
      float val = acc[v];
      if (Z) val += Z[o];
      Y[o] = val;
    }
  }
}

// out = dinv^2 * h + bias   (self-loop + bias; fully initializes out); float4
__global__ void k_prop_init(const float* __restrict__ h, const float* __restrict__ dinv,
                            const float* __restrict__ bias, float* __restrict__ out, int n) {
  int i = blockIdx.x * blockDim.x + threadIdx.x;     // quads
  if (i < n * (FDIM / 4)) {
    int node = i >> 5, q = (i & 31) << 2;
    float d = dinv[node]; d *= d;
    const float4 hv = *(const float4*)&h[(size_t)node * FDIM + q];
    const float4 bv = *(const float4*)&bias[q];
    float4 o = make_float4(d * hv.x + bv.x, d * hv.y + bv.y,
                           d * hv.z + bv.z, d * hv.w + bv.w);
    *(float4*)&out[(size_t)node * FDIM + q] = o;
  }
}

// out[rcv[e]] += dinv[snd]*dinv[rcv] * h[snd]   32 lanes x float4 per edge
__global__ void k_prop_edges(const float* __restrict__ h, const int* __restrict__ snd,
                             const int* __restrict__ rcv, const float* __restrict__ dinv,
                             float* __restrict__ out, int E) {
  long long i = (long long)blockIdx.x * blockDim.x + threadIdx.x;
  if (i < (long long)E * (FDIM / 4)) {
    int e = (int)(i >> 5), q = ((int)i & 31) << 2;
    int s = snd[e], r = rcv[e];
    float nm = dinv[s] * dinv[r];
    const float4 hv = *(const float4*)&h[(size_t)s * FDIM + q];
    float* o = &out[(size_t)r * FDIM + q];
    atomicAdd(o + 0, nm * hv.x);
    atomicAdd(o + 1, nm * hv.y);
    atomicAdd(o + 2, nm * hv.z);
    atomicAdd(o + 3, nm * hv.w);
  }
}

// dst = 0.5*(a + b); float4 (dst may alias a)
__global__ void k_avg(float* __restrict__ dst, const float* __restrict__ a,
                      const float* __restrict__ b, int nq) {
  int i = blockIdx.x * blockDim.x + threadIdx.x;
  if (i < nq) {
    const float4 av = ((const float4*)a)[i];
    const float4 bv = ((const float4*)b)[i];
    ((float4*)dst)[i] = make_float4(0.5f * (av.x + bv.x), 0.5f * (av.y + bv.y),
                                    0.5f * (av.z + bv.z), 0.5f * (av.w + bv.w));
  }
}

// c = alpha[ch]*c + 0.5*(h + p)    (final smooth iter fused with alpha residual)
__global__ void k_alpha_avg(float* __restrict__ c, const float* __restrict__ h,
                            const float* __restrict__ p, const float* __restrict__ alpha,
                            int nq) {
  int i = blockIdx.x * blockDim.x + threadIdx.x;
  if (i < nq) {
    int q = (i & 31) << 2;
    const float4 av = *(const float4*)&alpha[q];
    const float4 hv = ((const float4*)h)[i];
    const float4 pv = ((const float4*)p)[i];
    float4 cv = ((float4*)c)[i];
    cv.x = av.x * cv.x + 0.5f * (hv.x + pv.x);
    cv.y = av.y * cv.y + 0.5f * (hv.y + pv.y);
    cv.z = av.z * cv.z + 0.5f * (hv.z + pv.z);
    cv.w = av.w * cv.w + 0.5f * (hv.w + pv.w);
    ((float4*)c)[i] = cv;
  }
}

__global__ void k_concat(const float* __restrict__ c0, const float* __restrict__ c1,
                         float* __restrict__ out, int n) {
  int i = blockIdx.x * blockDim.x + threadIdx.x;     // quads of the [N,256] output
  if (i < n * (2 * FDIM / 4)) {
    int node = i >> 6, f = (i & 63) << 2;
    float4 v = (f < FDIM) ? *(const float4*)&c0[(size_t)node * FDIM + f]
                          : *(const float4*)&c1[(size_t)node * FDIM + (f - FDIM)];
    *(float4*)&out[(size_t)node * 2 * FDIM + f] = v;
  }
}

extern "C" void kernel_launch(void* const* d_in, const int* in_sizes, int n_in,
                              void* d_out, int out_size, void* d_ws, size_t ws_size,
                              hipStream_t stream) {
  const float* x   = (const float*)d_in[0];
  const int*   ei  = (const int*)d_in[1];
  const float* Wp  = (const float*)d_in[2];
  const float* bp  = (const float*)d_in[3];
  const float* al0 = (const float*)d_in[4];
  const float* al1 = (const float*)d_in[5];
  const float* W0  = (const float*)d_in[6];
  const float* b0  = (const float*)d_in[7];
  const float* W1  = (const float*)d_in[8];
  const float* b1  = (const float*)d_in[9];
  const float* Wdn = (const float*)d_in[10];
  const float* Wup = (const float*)d_in[11];
  float* out = (float*)d_out;

  const int N = in_sizes[0] / FDIM;
  const int E = in_sizes[1] / 2;
  const int S = in_sizes[4] / FDIM;   // alpha0: (S,1,128)
  const int* snd = ei;
  const int* rcv = ei + E;

  // workspace carve-out (256B aligned)
  char* ws = (char*)d_ws;
  size_t off = 0;
  auto take = [&](size_t bytes) -> void* {
    void* p = (void*)(ws + off);
    off = (off + bytes + 255) & ~(size_t)255;
    return p;
  };
  const size_t MAT = (size_t)N * FDIM * sizeof(float);
  float*  dinv = (float*)take((size_t)N * sizeof(float));
  __bf16* wbf  = (__bf16*)take(WTILE_BYTES);  // bf16 transposed W scratch
  float*  xp = (float*)take(MAT);  // x after gcn_pca
  float*  c0 = (float*)take(MAT);
  float*  c1 = (float*)take(MAT);
  float*  t  = (float*)take(MAT);  // GEMM staging
  float*  g  = (float*)take(MAT);  // gconv working buffer
  float*  p  = (float*)take(MAT);  // propagation scratch
  (void)ws_size; (void)n_in; (void)out_size;

  const int TB = 256;
  const int nq = N * (FDIM / 4);              // float4 count of a feature matrix
  dim3 gN((N + TB - 1) / TB), gE((E + TB - 1) / TB);
  dim3 gNQ((nq + TB - 1) / TB);
  long long eq = (long long)E * (FDIM / 4);
  dim3 gEQ((unsigned)((eq + TB - 1) / TB));
  dim3 gGemm((N + 15) / 16);
  dim3 gW((FDIM * FDIM + TB - 1) / TB);
  dim3 gC((N * (2 * FDIM / 4) + TB - 1) / TB);

  // symmetric normalization D^-1/2 (A + I) D^-1/2
  k_zero<<<gN, TB, 0, stream>>>(dinv, N);
  k_deg<<<gE, TB, 0, stream>>>(snd, dinv, E);
  k_dinv<<<gN, TB, 0, stream>>>(dinv, N);

  auto gemm = [&](const float* A, const float* Wm, const float* Zadd, float* Yo) {
    k_wconv<<<gW, TB, 0, stream>>>(Wm, wbf);           // W -> bf16 transposed (once)
    k_gemm_wmma<<<gGemm, TB, 0, stream>>>(A, wbf, Zadd, Yo, N);
  };
  auto prop = [&](const float* h, const float* bias) {  // p = prop(h) + bias
    k_prop_init<<<gNQ, TB, 0, stream>>>(h, dinv, bias, p, N);
    k_prop_edges<<<gEQ, TB, 0, stream>>>(h, snd, rcv, dinv, p, E);
  };
  auto smooth = [&](float* h, const float* bias, int iters) {
    for (int it = 0; it < iters; ++it) {                // h = 0.5*(h + prop(h))
      prop(h, bias);
      k_avg<<<gNQ, TB, 0, stream>>>(h, h, p, nq);
    }
  };
  // two smoothing iters; second fused with c = alpha*c + result
  auto smooth2_alpha = [&](float* h, const float* bias, float* c, const float* alpha) {
    prop(h, bias);
    k_avg<<<gNQ, TB, 0, stream>>>(h, h, p, nq);
    prop(h, bias);
    k_alpha_avg<<<gNQ, TB, 0, stream>>>(c, h, p, alpha, nq);
  };

  // gcn_pca: x = _gconv(x, Wp, bp, c=1)
  gemm(x, Wp, nullptr, xp);
  smooth(xp, bp, 1);

  for (int i = 0; i < S; ++i) {
    const float* W0i = W0  + (size_t)i * FDIM * FDIM;
    const float* W1i = W1  + (size_t)i * FDIM * FDIM;
    const float* Wdi = Wdn + (size_t)i * FDIM * FDIM;
    const float* Wui = Wup + (size_t)i * FDIM * FDIM;
    const float* b0i = b0 + (size_t)i * FDIM;
    const float* b1i = b1 + (size_t)i * FDIM;
    if (i == 0) {
      gemm(xp, W0i, nullptr, c0);                 // c0 = gconv(x, W0[0])
      smooth(c0, b0i, 2);
      gemm(c0, Wdi, nullptr, t);                  // t = c0 @ Wdown[0]
      gemm(t, W1i, nullptr, c1);                  // c1 = gconv(t, W1[0])
      smooth(c1, b1i, 2);
    } else {
      gemm(c1, Wui, xp, t);                       // t = c1 @ Wup[i] + x
      gemm(t, W0i, nullptr, g);                   // g = t @ W0[i]
      smooth2_alpha(g, b0i, c0, al0 + (size_t)i * FDIM);   // c0 = a0*c0 + gconv
      gemm(c0, Wdi, nullptr, t);                  // t = c0 @ Wdown[i]
      gemm(t, W1i, nullptr, g);                   // g = t @ W1[i]
      smooth2_alpha(g, b1i, c1, al1 + (size_t)i * FDIM);   // c1 = a1*c1 + gconv
    }
  }

  k_concat<<<gC, TB, 0, stream>>>(c0, c1, out, N);
}
